// Continuous_Embedding_29102698398139
// MI455X (gfx1250) — matrix operand order repaired and checked by reading, not compiled, not run on hardware
//
#include <hip/hip_runtime.h>
#include <hip/hip_fp16.h>

// MI455X / gfx1250, wave32. Matrix math via v_wmma_f32_16x16x32_f16.
//
// out[r][:] = MLP(feat(x[r])), feat = [embd[ic], embd[ic+1], dist], HID=101,
// OUT=50; OOB rows overridden with embd_oob. ~32 GFLOP vs ~214 MB traffic:
// f16 WMMA (K=32/instr) + fp32 accumulate puts this near the HBM roof.
// GELU via gfx1250's hardware v_tanh_f32 (tanh-form GELU == ~1e-3-accurate
// erf-GELU, same order as the f16 operand quantization); branchless A&S erf
// as fallback if the builtin is unavailable.

#define HIDW   101
#define OUTW   50
#define NBND   50
#define EMLEN  51
#define EMD    50
#define N1T    7      // ceil(101/16) GEMM1 n-tiles
#define N2T    4      // ceil(50/16)  GEMM2 n-tiles
#define KCH    4      // ceil(101/32) K chunks (pad to 128)

typedef __attribute__((ext_vector_type(16))) _Float16     v16h;
typedef __attribute__((ext_vector_type(8)))  float        v8f;
typedef __attribute__((ext_vector_type(4)))  unsigned int u32x4;

union Frag {
  v16h v;
  unsigned int u[8];
  u32x4 q[2];
};

__device__ __forceinline__ unsigned int packf16(float a, float b) {
  union { _Float16 h[2]; unsigned int u; } p;
  p.h[0] = (_Float16)a; p.h[1] = (_Float16)b;
  return p.u;
}

#if __has_builtin(__builtin_amdgcn_tanhf)
// 6-op GELU using gfx1250 hardware tanh:
// gelu(v) ~= 0.5*v*(1 + tanh(sqrt(2/pi)*(v + 0.044715*v^3)))
__device__ __forceinline__ float gelu_erf(float v) {
  float v2    = v * v;
  float inner = (0.79788456080286535588f * v) * __builtin_fmaf(0.044715f, v2, 1.0f);
  float th    = __builtin_amdgcn_tanhf(inner);   // v_tanh_f32
  float hv    = 0.5f * v;
  return __builtin_fmaf(hv, th, hv);
}
#else
// Branchless exact-erf GELU (A&S 7.1.26): 1x v_rcp_f32, 1x v_exp_f32, ~8 FMA.
__device__ __forceinline__ float gelu_erf(float v) {
  float z  = v * 0.70710678118654752440f;
  float az = __builtin_fabsf(z);
  float t  = __builtin_amdgcn_rcpf(__builtin_fmaf(0.3275911f, az, 1.0f));
  float p  = __builtin_fmaf(t, 1.061405429f, -1.453152027f);
  p = __builtin_fmaf(t, p, 1.421413741f);
  p = __builtin_fmaf(t, p, -0.284496736f);
  p = __builtin_fmaf(t, p, 0.254829592f);
  p = p * t;
  float ex = __expf(-az * az);
  float er = 1.0f - p * ex;
  er = __builtin_copysignf(er, z);
  return 0.5f * v * (1.0f + er);
}
#endif

__global__ __launch_bounds__(128) void Continuous_Embedding_29102698398139_kernel(
    const float* __restrict__ X,   const float* __restrict__ Bnd,
    const float* __restrict__ Emb, const float* __restrict__ Oob,
    const float* __restrict__ W1,  const float* __restrict__ B1,
    const float* __restrict__ W2,  const float* __restrict__ B2,
    float* __restrict__ Out, int nTiles) {
  // ---- LDS: weights pre-swizzled into WMMA B-fragment layout (f16) ----
  // B frag layout (16-bit B, 32x16): lane l holds N=l&15, K = 8*(l>>4)+(j<4 ? 2j : 16+2(j-4))+{0,1}
  // -> per lane 16 K halfs = 32B, stored fragment-major.
  __shared__ unsigned int   sW1[N1T * KCH * 256]; // 28 frags * 1KB = 28672 B
  __shared__ unsigned int   sW2[KCH * N2T * 256]; // 16 frags * 1KB = 16384 B
  __shared__ unsigned int   sEmbd[1280];          // 51*50 halfs contiguous (5.1KB)
  __shared__ float          sB1[112];             // padded with zeros
  __shared__ float          sB2[64];
  __shared__ float          sBnd[NBND];
  __shared__ float          sOob[100];
  __shared__ unsigned short sH[4][512];           // per-wave h chunk: 16 rows x 32 cols f16

  const int tid = threadIdx.x;

  // W1 fragments: B[k][n] = w1[n][k]  (H = feat @ W1^T)
  for (int g = tid; g < N1T * KCH * 256; g += 128) {
    int f = g >> 8, rem = g & 255, l = rem >> 3, j = rem & 7;
    int nt = f >> 2, kc = f & 3;
    int n = nt * 16 + (l & 15);
    int K = kc * 32 + (l >> 4) * 8 + ((j < 4) ? (2 * j) : (16 + 2 * (j - 4)));
    float a = (n < HIDW && K     < HIDW) ? W1[n * HIDW + K]     : 0.0f;
    float b = (n < HIDW && K + 1 < HIDW) ? W1[n * HIDW + K + 1] : 0.0f;
    sW1[g] = packf16(a, b);
  }
  // W2 fragments: B[k][n] = w2[n][k]  (out = h @ W2^T); layout [kchunk][ntile]
  for (int g = tid; g < KCH * N2T * 256; g += 128) {
    int f = g >> 8, rem = g & 255, l = rem >> 3, j = rem & 7;
    int kc = f >> 2, nt = f & 3;
    int n = nt * 16 + (l & 15);
    int K = kc * 32 + (l >> 4) * 8 + ((j < 4) ? (2 * j) : (16 + 2 * (j - 4)));
    float a = (n < OUTW && K     < HIDW) ? W2[n * HIDW + K]     : 0.0f;
    float b = (n < OUTW && K + 1 < HIDW) ? W2[n * HIDW + K + 1] : 0.0f;
    sW2[g] = packf16(a, b);
  }
  // Embeddings as f16, rows contiguous (stride 50 halfs) so that
  // feat[k] = sEmbd_half[ic*50 + k] for k < 100.
  for (int g = tid; g < 1280; g += 128) {
    int i0 = 2 * g, i1 = 2 * g + 1;
    float a = (i0 < EMLEN * EMD) ? Emb[i0] : 0.0f;
    float b = (i1 < EMLEN * EMD) ? Emb[i1] : 0.0f;
    sEmbd[g] = packf16(a, b);
  }
  for (int i = tid; i < 112; i += 128) sB1[i] = (i < HIDW) ? B1[i] : 0.0f;
  for (int i = tid; i < 64;  i += 128) sB2[i] = (i < OUTW) ? B2[i] : 0.0f;
  for (int i = tid; i < NBND; i += 128) sBnd[i] = Bnd[i];
  for (int i = tid; i < 100; i += 128) sOob[i] = Oob[i];
  __syncthreads();

  const int lane = tid & 31;
  const int wave = tid >> 5;
  const int m    = lane & 15;   // row within M-tile (A layout), column within n-tile (C layout)
  const int hh   = lane >> 4;   // lane half-select
  unsigned short* __restrict__ hbuf = sH[wave];

  const int wavesTotal = gridDim.x * 4;
  for (int t = blockIdx.x * 4 + wave; t < nTiles; t += wavesTotal) {
    const int rowBase = t * 16;
    if (t + wavesTotal < nTiles)
      __builtin_prefetch(&X[(t + wavesTotal) * 16 + m], 0, 3); // global_prefetch_b8

    // ---- scalar prologue: searchsorted(left) + interpolation ----
    float x = X[rowBase + m];
    int idx = 0;                      // = count(bnd < x), in [0,50]
#pragma unroll
    for (int step = 32; step >= 1; step >>= 1) {
      int tt = idx + step;
      if (tt <= NBND && sBnd[tt - 1] < x) idx = tt;
    }
    float lov = sBnd[idx > 0 ? idx - 1 : 0];
    float hiv = sBnd[idx < NBND ? idx : NBND - 1];
    float rng = hiv - lov;
    // result feeds an f16 operand; 1-ulp v_rcp_f32 beats the IEEE div sequence
    float dist = (x - lov) * __builtin_amdgcn_rcpf((rng == 0.0f) ? 1.0f : rng);
    int ic = (idx < 1) ? 1 : ((idx > EMLEN - 2) ? (EMLEN - 2) : idx);
    union { _Float16 h; unsigned short s; } dh; dh.h = (_Float16)dist;

    // broadcast idx for all 16 rows while flow is convergent (shfl from
    // inactive lanes returns 0 -> must not live under the n<50 predicate)
    int idxRow[8];
#pragma unroll
    for (int r = 0; r < 8; ++r) idxRow[r] = __shfl(idx, r + 8 * hh, 32);

    // ---- build A fragments (feat, 16x32 f16 per K-chunk, ISA A-layout) ----
    // lane holds row m; VGPR j: K = kc*32 + 8*hh + (j<4 ? 2j : 16+2(j-4))
    Frag fa[KCH];
#pragma unroll
    for (int kc = 0; kc < KCH; ++kc) {
#pragma unroll
      for (int j = 0; j < 8; ++j) {
        int K = kc * 32 + hh * 8 + ((j < 4) ? (2 * j) : (16 + 2 * (j - 4)));
        int Kc = (K < 98) ? K : 98;                       // clamp gather addr
        unsigned int w = sEmbd[ic * 25 + (Kc >> 1)];      // halfs (K,K+1), K<100
        if (K == 100) w = (unsigned int)dh.s;             // (dist, 0)
        else if (K > 100) w = 0u;
        fa[kc].u[j] = w;
      }
    }

    // ---- fused GEMM1 -> GELU -> GEMM2, streamed per 32-col h chunk ----
    v8f acc2[N2T] = {};
#pragma unroll
    for (int c = 0; c < 4; ++c) {
#pragma unroll
      for (int d = 0; d < 2; ++d) {
        int nt = 2 * c + d;
        if (nt < N1T) {
          v8f acc1 = {};
#pragma unroll
          for (int kc = 0; kc < KCH; ++kc) {
            Frag bf;
            const u32x4* bp = (const u32x4*)&sW1[(nt * 4 + kc) * 256 + lane * 8];
            bf.q[0] = bp[0]; bf.q[1] = bp[1];
            acc1 = __builtin_amdgcn_wmma_f32_16x16x32_f16(
                false, fa[kc].v, false, bf.v, (short)0, acc1, false, false);
          }
          float bias = sB1[nt * 16 + m];
#pragma unroll
          for (int r = 0; r < 8; ++r) {
            float v = gelu_erf(acc1[r] + bias);           // gelu(0)=0 keeps pads clean
            union { _Float16 h; unsigned short s; } cv; cv.h = (_Float16)v;
            hbuf[(r + 8 * hh) * 32 + d * 16 + m] = cv.s;  // row-major [16][32]
          }
        } else { // n-tile 7: cols 112..127 are pure zero padding
#pragma unroll
          for (int r = 0; r < 8; ++r)
            hbuf[(r + 8 * hh) * 32 + 16 + m] = 0;
        }
      }
      // GEMM2 A fragment straight out of the h chunk (same-wave LDS is in-order)
      Frag ha;
      const u32x4* hp = (const u32x4*)hbuf;
      ha.q[0] = hp[m * 4 + hh];       // halfs  8*hh .. 8*hh+7   of row m
      ha.q[1] = hp[m * 4 + 2 + hh];   // halfs 16+8*hh .. 23+8*hh
#pragma unroll
      for (int nt2 = 0; nt2 < N2T; ++nt2) {
        Frag bf;
        const u32x4* bp = (const u32x4*)&sW2[(c * 4 + nt2) * 256 + lane * 8];
        bf.q[0] = bp[0]; bf.q[1] = bp[1];
        acc2[nt2] = __builtin_amdgcn_wmma_f32_16x16x32_f16(
            false, ha.v, false, bf.v, (short)0, acc2[nt2], false, false);
      }
    }

    // ---- bias + OOB override + coalesced writeback ----
#pragma unroll
    for (int nt = 0; nt < N2T; ++nt) {
      int n = nt * 16 + m;
      if (n < OUTW) {
        float o0 = sOob[n], o1 = sOob[OUTW + n], bb = sB2[n];
#pragma unroll
        for (int r = 0; r < 8; ++r) {
          int rowm = r + 8 * hh;
          int ir = idxRow[r];
          float v = acc2[nt][r] + bb;
          v = (ir == 0) ? o0 : ((ir == EMLEN - 1) ? o1 : v);
          Out[(size_t)(rowBase + rowm) * OUTW + n] = v;
        }
      }
    }
  }
}

extern "C" void kernel_launch(void* const* d_in, const int* in_sizes, int n_in,
                              void* d_out, int out_size, void* d_ws, size_t ws_size,
                              hipStream_t stream) {
  const float* X   = (const float*)d_in[0];
  const float* Bnd = (const float*)d_in[1];
  const float* Emb = (const float*)d_in[2];
  const float* Oob = (const float*)d_in[3];
  const float* W1  = (const float*)d_in[4];
  const float* B1  = (const float*)d_in[5];
  const float* W2  = (const float*)d_in[6];
  const float* B2  = (const float*)d_in[7];
  float* Out = (float*)d_out;

  int rows   = in_sizes[0];     // B*N = 1,048,576
  int nTiles = rows >> 4;       // 16-row M-tiles

  dim3 grid(2048), block(128);  // 8192 waves -> 8 tiles/wave, weights L2-resident
  Continuous_Embedding_29102698398139_kernel<<<grid, block, 0, stream>>>(
      X, Bnd, Emb, Oob, W1, B1, W2, B2, Out, nTiles);
}
